// ResBlock_64785286693636
// MI455X (gfx1250) — matrix operand order repaired and checked by reading, not compiled, not run on hardware
//
#include <hip/hip_runtime.h>
#include <stdint.h>

#define OFFS 27
#define C_IN 96
#define C_MID 192

typedef __attribute__((ext_vector_type(16))) __bf16 v16bf;
typedef __attribute__((ext_vector_type(8)))  float  v8f;

union FragU { v16bf v; uint4 q[2]; };

__device__ __forceinline__ unsigned short f2bf(float f) {
  unsigned int u = __float_as_uint(f);
  return (unsigned short)((u + 0x7fffu + ((u >> 16) & 1u)) >> 16);
}

__global__ void zero_f32(float* __restrict__ p, int count) {
  int i = blockIdx.x * blockDim.x + threadIdx.x;
  if (i < count) p[i] = 0.0f;
}

__global__ void cvt_bf16(const float* __restrict__ src, unsigned short* __restrict__ dst,
                         size_t total) {
  size_t i = (size_t)blockIdx.x * blockDim.x + threadIdx.x;
  if (i < total) dst[i] = f2bf(src[i]);
}

// Swizzle W[o][K][col] (f32) into per-wave B fragments (bf16):
// element e = ((o*KB+kb)*NB+nb)*512 + lane*16 + h
// lane<16 : column n0+lane, K = kb*32 + h        (halfs 0..15 = K 0..15)
// lane>=16: column n0+lane-16, K = kb*32 + 16 + h
template<int CIN, int COUT>
__global__ void prep_w(const float* __restrict__ W, unsigned short* __restrict__ Wsw) {
  constexpr int KB = CIN / 32, NB = COUT / 16;
  int e = blockIdx.x * blockDim.x + threadIdx.x;
  if (e >= OFFS * KB * NB * 512) return;
  int h    = e & 15;
  int lane = (e >> 4) & 31;
  int f    = e >> 9;
  int nb   = f % NB;
  int t    = f / NB;
  int kb   = t % KB;
  int o    = t / KB;
  int K    = kb * 32 + ((lane >> 4) << 4) + h;
  int col  = nb * 16 + (lane & 15);
  Wsw[e] = f2bf(W[((size_t)o * CIN + K) * COUT + col]);
}

// Sparse conv via WMMA bf16: out[N,COUT] f32 = sum_o gather(Xh, idx[o], mask[o]) @ W[o]
// Also accumulates per-channel sum / sum-of-squares for BatchNorm (f32 atomics).
//
// Block = 256 threads = 8 waves = 4 row-groups x 2 column-halves; 128 rows/block.
// Each wave owns 32 rows (two 16-row A sets) x COUT/2 columns, so every LDS
// B-fragment read feeds TWO wmma ops (halves LDS bandwidth per FLOP).
// Weights are DMA'd into a DOUBLE-BUFFERED LDS stage with the CDNA5 async path
// (GLOBAL_LOAD_ASYNC_TO_LDS_B128 / ASYNCcnt), so offset o+1's 36KB stage runs
// underneath offset o's WMMA work.
template<int CIN, int COUT>
__global__ __launch_bounds__(256)
void spconv_wmma(const unsigned short* __restrict__ Xh,   // [N, CIN] bf16
                 const int* __restrict__ idx,             // [OFFS, N]
                 const int* __restrict__ mask,            // [OFFS, N]
                 const unsigned short* __restrict__ Wsw,  // swizzled bf16 fragments
                 float* __restrict__ out,                 // [N, COUT] f32
                 float* __restrict__ sum, float* __restrict__ sumsq,
                 int n)
{
  constexpr int KB = CIN / 32, NB = COUT / 16, NBW = NB / 2;
  constexpr int STAGE_HALFS = KB * NB * 512;              // 36 KB per offset
  static_assert(KB * NB == 36, "stage loop assumes 36KB of weights per offset");
  __shared__ unsigned short lds[2 * STAGE_HALFS];         // 72 KB double buffer

  const int lane   = threadIdx.x & 31;
  const int wave   = threadIdx.x >> 5;
  const int rowgrp = wave >> 1;                           // 0..3
  const int nb0    = (wave & 1) * NBW;                    // column half
  const int row0   = (blockIdx.x * 4 + rowgrp) * 32;

  const int r0 = row0 + (lane & 15);
  const int r1 = r0 + 16;
  const bool v0 = r0 < n;
  const bool v1 = r1 < n;
  const int rc0 = v0 ? r0 : (n - 1);
  const int rc1 = v1 ? r1 : (n - 1);
  const int hsel = (lane >> 4) << 4;                      // A-layout half-select: 0 or 16

  v8f acc[2][NBW] = {};
  const uint4 zq = make_uint4(0u, 0u, 0u, 0u);

  const unsigned ldsBase = (unsigned)(uintptr_t)(&lds[0]);

  // Issue one offset's 36KB weight DMA into LDS buffer `buf` (256 thr x 9 x b128).
  auto stage = [&](int o, unsigned bufBytes) {
    const char* gsrc = (const char*)(Wsw + (size_t)o * STAGE_HALFS);
    #pragma unroll
    for (int t = 0; t < 9; ++t) {
      unsigned la = ldsBase + bufBytes + threadIdx.x * 16 + t * 4096;
      unsigned long long ga =
          (unsigned long long)(uintptr_t)(gsrc + threadIdx.x * 16 + t * 4096);
      asm volatile("global_load_async_to_lds_b128 %0, %1, off"
                   :: "v"(la), "v"(ga) : "memory");
    }
  };

  stage(0, 0u);                                           // prologue prefetch

  for (int o = 0; o < OFFS; ++o) {
    const unsigned curBytes = (o & 1) ? (unsigned)(STAGE_HALFS * 2) : 0u;
    const unsigned short* curW = lds + (o & 1) * STAGE_HALFS;

    // Gather metadata for this offset: independent global loads, issue early.
    const int iv0 = idx[(size_t)o * n + rc0];
    const int iv1 = idx[(size_t)o * n + rc1];
    const int mv0 = v0 ? mask[(size_t)o * n + rc0] : 0;
    const int mv1 = v1 ? mask[(size_t)o * n + rc1] : 0;

    // My DMA into the current buffer is done; barrier publishes everyone's DMA
    // and guarantees all waves finished reading the other buffer.
    asm volatile("s_wait_asynccnt 0x0" ::: "memory");
    __syncthreads();

    if (o + 1 < OFFS)                                     // overlap next stage
      stage(o + 1, curBytes ? 0u : (unsigned)(STAGE_HALFS * 2));

    const unsigned short* __restrict__ xr0 = Xh + (size_t)iv0 * CIN;
    const unsigned short* __restrict__ xr1 = Xh + (size_t)iv1 * CIN;

    #pragma unroll
    for (int kb = 0; kb < KB; ++kb) {
      // Gather the two A fragments (16-bit A 16x32 layout): 2 b128 chunks/lane.
      FragU a0, a1;
      {
        const uint4* ap = (const uint4*)(xr0 + kb * 32 + hsel);
        uint4 q0 = ap[0], q1 = ap[1];
        a0.q[0] = (mv0 > 0) ? q0 : zq;
        a0.q[1] = (mv0 > 0) ? q1 : zq;
      }
      {
        const uint4* ap = (const uint4*)(xr1 + kb * 32 + hsel);
        uint4 q0 = ap[0], q1 = ap[1];
        a1.q[0] = (mv1 > 0) ? q0 : zq;
        a1.q[1] = (mv1 > 0) ? q1 : zq;
      }

      #pragma unroll
      for (int j = 0; j < NBW; ++j) {
        FragU b;
        const uint4* bp =
            (const uint4*)(curW + ((kb * NB + nb0 + j) << 9) + (lane << 4));
        b.q[0] = bp[0];
        b.q[1] = bp[1];
        acc[0][j] = __builtin_amdgcn_wmma_f32_16x16x32_bf16(
            false, a0.v, false, b.v, (short)0, acc[0][j], false, false);
        acc[1][j] = __builtin_amdgcn_wmma_f32_16x16x32_bf16(
            false, a1.v, false, b.v, (short)0, acc[1][j], false, false);
      }
    }
  }

  // Epilogue: store f32 tiles + per-channel BN statistics.
  // C layout: lane<16 -> col=lane, rows p; lane>=16 -> col=lane-16, rows p+8.
  const int rowadd = (lane >> 4) << 3;
  #pragma unroll
  for (int rs = 0; rs < 2; ++rs) {
    #pragma unroll
    for (int j = 0; j < NBW; ++j) {
      const int col = (nb0 + j) * 16 + (lane & 15);
      float s = 0.f, s2 = 0.f;
      #pragma unroll
      for (int p = 0; p < 8; ++p) {
        float v = acc[rs][j][p];
        s += v;
        s2 += v * v;
        int rr = row0 + rs * 16 + rowadd + p;
        if (rr < n) out[(size_t)rr * COUT + col] = v;
      }
      atomicAdd(&sum[col], s);
      atomicAdd(&sumsq[col], s2);
    }
  }
}

__global__ void bn_finalize(const float* __restrict__ sum, const float* __restrict__ sq,
                            const float* __restrict__ g, const float* __restrict__ b,
                            float* __restrict__ scale, float* __restrict__ shift,
                            int C, float invN)
{
  int c = blockIdx.x * blockDim.x + threadIdx.x;
  if (c < C) {
    float mu  = sum[c] * invN;
    float var = sq[c] * invN - mu * mu;
    float sc  = g[c] * rsqrtf(var + 1e-5f);
    scale[c] = sc;
    shift[c] = b[c] - mu * sc;
  }
}

__global__ void bn_relu_cvt(const float* __restrict__ h, const float* __restrict__ scale,
                            const float* __restrict__ shift, unsigned short* __restrict__ outb,
                            size_t total, int C)
{
  size_t i = (size_t)blockIdx.x * blockDim.x + threadIdx.x;
  if (i < total) {
    int c = (int)(i % C);
    float v = fmaf(h[i], scale[c], shift[c]);
    outb[i] = f2bf(v > 0.f ? v : 0.f);
  }
}

__global__ void bn_residual_out(const float* __restrict__ h2, const float* __restrict__ x,
                                const float* __restrict__ scale, const float* __restrict__ shift,
                                float* __restrict__ out, size_t total, int C)
{
  size_t i = (size_t)blockIdx.x * blockDim.x + threadIdx.x;
  if (i < total) {
    int c = (int)(i % C);
    float v = fmaf(h2[i], scale[c], shift[c]) + x[i];
    out[i] = v > 0.f ? v : 0.f;
  }
}

extern "C" void kernel_launch(void* const* d_in, const int* in_sizes, int n_in,
                              void* d_out, int out_size, void* d_ws, size_t ws_size,
                              hipStream_t stream)
{
  const float* x   = (const float*)d_in[0];
  const int*   idx = (const int*)d_in[1];
  const int*   msk = (const int*)d_in[2];
  const float* W1  = (const float*)d_in[3];
  const float* g1  = (const float*)d_in[4];
  const float* b1  = (const float*)d_in[5];
  const float* W2  = (const float*)d_in[6];
  const float* g2  = (const float*)d_in[7];
  const float* b2  = (const float*)d_in[8];
  float* out = (float*)d_out;
  const int n = in_sizes[0] / C_IN;

  // Workspace carve-up (256B aligned).
  char* w = (char*)d_ws;
  size_t off = 0;
  auto alloc = [&](size_t bytes) -> char* {
    char* p = w + off;
    off += (bytes + 255) & ~(size_t)255;
    return p;
  };
  unsigned short* xh  = (unsigned short*)alloc((size_t)n * C_IN  * 2);  // x  bf16
  unsigned short* hb  = (unsigned short*)alloc((size_t)n * C_MID * 2);  // relu(bn1) bf16
  float*          h1  = (float*)alloc((size_t)n * C_MID * 4);           // conv1 f32
  float*          h2  = (float*)alloc((size_t)n * C_IN  * 4);           // conv2 f32
  unsigned short* w1s = (unsigned short*)alloc((size_t)OFFS * C_IN  * C_MID * 2);
  unsigned short* w2s = (unsigned short*)alloc((size_t)OFFS * C_MID * C_IN  * 2);
  float* stats = (float*)alloc((size_t)(4 * C_MID + 4 * C_IN) * 4);
  float* sum1 = stats;            float* sq1 = sum1 + C_MID;
  float* sum2 = sq1 + C_MID;      float* sq2 = sum2 + C_IN;
  float* sc1  = sq2 + C_IN;       float* sh1 = sc1 + C_MID;
  float* sc2  = sh1 + C_MID;      float* sh2 = sc2 + C_IN;

  const int statN = 2 * (C_MID + C_IN);
  zero_f32<<<(statN + 255) / 256, 256, 0, stream>>>(stats, statN);

  { int tot = OFFS * (C_IN / 32) * (C_MID / 16) * 512;
    prep_w<C_IN, C_MID><<<(tot + 255) / 256, 256, 0, stream>>>(W1, w1s); }
  { int tot = OFFS * (C_MID / 32) * (C_IN / 16) * 512;
    prep_w<C_MID, C_IN><<<(tot + 255) / 256, 256, 0, stream>>>(W2, w2s); }
  { size_t tot = (size_t)n * C_IN;
    cvt_bf16<<<(unsigned)((tot + 255) / 256), 256, 0, stream>>>(x, xh, tot); }

  const int cblocks = (n + 127) / 128;   // 128 rows per block

  spconv_wmma<C_IN, C_MID><<<cblocks, 256, 0, stream>>>(xh, idx, msk, w1s, h1, sum1, sq1, n);
  bn_finalize<<<1, 256, 0, stream>>>(sum1, sq1, g1, b1, sc1, sh1, C_MID, 1.0f / (float)n);
  { size_t tot = (size_t)n * C_MID;
    bn_relu_cvt<<<(unsigned)((tot + 255) / 256), 256, 0, stream>>>(h1, sc1, sh1, hb, tot, C_MID); }

  spconv_wmma<C_MID, C_IN><<<cblocks, 256, 0, stream>>>(hb, idx, msk, w2s, h2, sum2, sq2, n);
  bn_finalize<<<1, 256, 0, stream>>>(sum2, sq2, g2, b2, sc2, sh2, C_IN, 1.0f / (float)n);
  { size_t tot = (size_t)n * C_IN;
    bn_residual_out<<<(unsigned)((tot + 255) / 256), 256, 0, stream>>>(h2, x, sc2, sh2, out, tot, C_IN); }
}
